// DMKT_37984690766520
// MI455X (gfx1250) — matrix operand order repaired and checked by reading, not compiled, not run on hardware
//
#include <hip/hip_runtime.h>
#include <hip/hip_bf16.h>
#include <math.h>

typedef __attribute__((ext_vector_type(16))) _Float16 v16h;
typedef __attribute__((ext_vector_type(8)))  _Float16 v8h;
typedef __attribute__((ext_vector_type(8)))  float    v8f;

union V16u { v16h v; v8h h[2]; };
union EAu  { unsigned u; _Float16 h[2]; };

// Problem dims
constexpr int Bn = 64, Tn = 200, LECn = 128;
constexpr int Cn = 128, Kn = 128, Vn = 128, Sn = 256;

// LDS strides
constexpr int HS = 136;   // halves per row for staged X (272B rows, 16B aligned)
constexpr int FS = 132;   // floats per row for lcw (528B rows, 16B aligned)
constexpr int EAS = 136;  // dwords per row for interleaved {e,a} pairs

// ---- dynamic LDS layout (bytes) ----
constexpr size_t SZFRAG  = 4 * 8 * 32 * 16 * sizeof(_Float16); // 32768, fragment-major weight
constexpr size_t OFF_WK  = 0;                                  // key^T  fragments
constexpr size_t OFF_WE  = OFF_WK + SZFRAG;                    // eraseW^T fragments
constexpr size_t OFF_WA  = OFF_WE + SZFRAG;                    // addW^T fragments
constexpr size_t OFF_X   = OFF_WA + SZFRAG;                    // staged L f16 [j][k], row-major
constexpr size_t OFF_EA  = OFF_X  + 128 * HS * sizeof(_Float16);   // packed {e,a} f16x2 [j][v]
constexpr size_t OFF_LCW = OFF_EA + 128 * EAS * 4;             // softmax scores f32 [j][c]
constexpr size_t OFF_QE  = OFF_LCW + 128 * FS * 4;
constexpr size_t OFF_QA  = OFF_QE  + 512;
constexpr size_t OFF_CW  = OFF_QA  + 512;
constexpr size_t OFF_EQ  = OFF_CW  + 512;
constexpr size_t OFF_AQ  = OFF_EQ  + 512;
constexpr size_t OFF_QR  = OFF_AQ  + 512;
constexpr size_t OFF_LR  = OFF_QR  + 512;
constexpr size_t OFF_LS  = OFF_LR  + 512;
constexpr size_t OFF_RED = OFF_LS  + 512;                      // 256 f32 tree-reduce scratch
constexpr size_t OFF_M   = OFF_RED + 1024;                     // 512 f32 summary input
constexpr size_t OFF_H   = OFF_M   + 2048;                     // 256 f32 hidden
constexpr size_t OFF_IDX = OFF_H   + 1024;                     // 128 int l_data indices
constexpr size_t SMEM_TOTAL = OFF_IDX + 512;                   // ~273 KB (< 320 KB WGP LDS)

// Element (k,n) inside a fragment-major weight block (for scalar matvec access).
__device__ __forceinline__ int fragOff(int k, int n) {
  return ((((k >> 5) << 3) + (n >> 4)) * 32 + (n & 15) + (k & 16)) * 16 + (k & 15);
}

// Load this wave's four A fragments (M-tile = wv): two contiguous 8-half runs each.
__device__ __forceinline__ void load_afrags(const _Float16* __restrict__ X,
                                            int wv, int lane, v16h afr[4]) {
  const int hi = lane >> 4, lrow = lane & 15;
  #pragma unroll
  for (int kb = 0; kb < 4; ++kb) {
    const _Float16* base = X + (wv * 16 + lrow) * HS + kb * 32 + hi * 16;
    V16u a;
    a.h[0] = *(const v8h*)(base);        // K = kbase .. kbase+7
    a.h[1] = *(const v8h*)(base + 8);    // K = kbase+8 .. kbase+15
    afr[kb] = a.v;
  }
}

// Scores = X[128x128] @ Wkey[128x128] -> raw f32 rows for softmax (no bias).
__device__ __forceinline__ void gemm_score(const _Float16* __restrict__ X,
                                           const _Float16* __restrict__ Wfrag,
                                           float* __restrict__ outF, int wv, int lane)
{
  const int hi = lane >> 4, lrow = lane & 15;
  const v16h* __restrict__ Wv = (const v16h*)Wfrag;
  v16h afr[4];
  load_afrags(X, wv, lane, afr);
  #pragma unroll 1
  for (int nt = 0; nt < 8; ++nt) {
    const int n = nt * 16 + lrow;
    v8f acc = {};
    #pragma unroll
    for (int kb = 0; kb < 4; ++kb) {
      const v16h bfr = Wv[(kb * 8 + nt) * 32 + lane];  // 2x ds_load_b128
      acc = __builtin_amdgcn_wmma_f32_16x16x32_f16(false, afr[kb], false, bfr,
                                                   (short)0, acc, false, false);
    }
    #pragma unroll
    for (int r = 0; r < 8; ++r)
      outF[(wv * 16 + r + hi * 8) * FS + n] = acc[r];  // no +0.0 bias add
  }
}

// Fused E/A GEMM: shares A fragments, two WMMA chains per K-block, packs
// {sigmoid(e), tanh(a)} into one dword -> single ds_store_b32 per element.
__device__ __forceinline__ void gemm_ea(const _Float16* __restrict__ X,
                                        const _Float16* __restrict__ WEf,
                                        const _Float16* __restrict__ WAf,
                                        const float* __restrict__ eb,
                                        const float* __restrict__ ab,
                                        _Float16* __restrict__ outEA, int wv, int lane)
{
  const int hi = lane >> 4, lrow = lane & 15;
  const v16h* __restrict__ WEv = (const v16h*)WEf;
  const v16h* __restrict__ WAv = (const v16h*)WAf;
  unsigned* __restrict__ pEA = (unsigned*)outEA;
  v16h afr[4];
  load_afrags(X, wv, lane, afr);
  #pragma unroll 1
  for (int nt = 0; nt < 8; ++nt) {
    const int n = nt * 16 + lrow;
    v8f accE = {}, accA = {};
    #pragma unroll
    for (int kb = 0; kb < 4; ++kb) {
      const v16h be = WEv[(kb * 8 + nt) * 32 + lane];
      const v16h ba = WAv[(kb * 8 + nt) * 32 + lane];
      accE = __builtin_amdgcn_wmma_f32_16x16x32_f16(false, afr[kb], false, be,
                                                    (short)0, accE, false, false);
      accA = __builtin_amdgcn_wmma_f32_16x16x32_f16(false, afr[kb], false, ba,
                                                    (short)0, accA, false, false);
    }
    const float ebn = eb[n], abn = ab[n];
    #pragma unroll
    for (int r = 0; r < 8; ++r) {
      const int mrow = wv * 16 + r + hi * 8;
      EAu ea;
      ea.h[0] = (_Float16)(1.0f / (1.0f + __expf(-(accE[r] + ebn))));
      ea.h[1] = (_Float16)tanhf(accA[r] + abn);
      pEA[mrow * EAS + n] = ea.u;                      // one ds_store_b32
    }
  }
}

__global__ void __launch_bounds__(256)
dmkt_kernel(const int* __restrict__ q_data, const int* __restrict__ qa_data,
            const int* __restrict__ l_data,
            const float* __restrict__ key_matrix, const float* __restrict__ value_init,
            const float* __restrict__ q_emb, const float* __restrict__ qa_emb,
            const float* __restrict__ l_emb,
            const float* __restrict__ erase_W, const float* __restrict__ erase_b,
            const float* __restrict__ add_W,   const float* __restrict__ add_b,
            const float* __restrict__ summary_W, const float* __restrict__ summary_b,
            const float* __restrict__ out_W,   const float* __restrict__ out_b,
            float* __restrict__ out)
{
  extern __shared__ char smem[];
  _Float16* sWK  = (_Float16*)(smem + OFF_WK);
  _Float16* sWE  = (_Float16*)(smem + OFF_WE);
  _Float16* sWA  = (_Float16*)(smem + OFF_WA);
  _Float16* sX   = (_Float16*)(smem + OFF_X);
  _Float16* sEA  = (_Float16*)(smem + OFF_EA);
  float*    sLcw = (float*)   (smem + OFF_LCW);
  float*    sQe  = (float*)   (smem + OFF_QE);
  float*    sQa  = (float*)   (smem + OFF_QA);
  float*    sCw  = (float*)   (smem + OFF_CW);
  float*    sEq  = (float*)   (smem + OFF_EQ);
  float*    sAq  = (float*)   (smem + OFF_AQ);
  float*    sQr  = (float*)   (smem + OFF_QR);
  float*    sLr  = (float*)   (smem + OFF_LR);
  float*    sLs  = (float*)   (smem + OFF_LS);
  float*    sRed = (float*)   (smem + OFF_RED);
  float*    sM   = (float*)   (smem + OFF_M);
  float*    sH   = (float*)   (smem + OFF_H);
  int*      sIdx = (int*)     (smem + OFF_IDX);

  const int b    = blockIdx.x;       // one batch element per workgroup
  const int tid  = threadIdx.x;
  const int wv   = tid >> 5;         // wave id 0..7 (wave32)
  const int lane = tid & 31;

  // Preload weights into fragment-major f16 layout (once).
  for (int idx = tid; idx < 4 * 8 * 32 * 16; idx += 256) {
    const int j  = idx & 15;
    const int ln = (idx >> 4) & 31;
    const int nt = (idx >> 9) & 7;
    const int kb = idx >> 12;
    const int n  = nt * 16 + (ln & 15);
    const int k  = kb * 32 + ((ln >> 4) & 1) * 16 + j;
    sWK[idx] = (_Float16)key_matrix[n * 128 + k];  // B[k][c] = key[c][k]
    sWE[idx] = (_Float16)erase_W[n * 128 + k];
    sWA[idx] = (_Float16)add_W[n * 128 + k];
  }

  // vm resident in VGPRs: wave wv owns rows c = 16*wv + r; lane owns v = 32*q + lane.
  float vm[16][4];
  #pragma unroll
  for (int r = 0; r < 16; ++r)
    #pragma unroll
    for (int q = 0; q < 4; ++q)
      vm[r][q] = value_init[(wv * 16 + r) * Vn + (q * 32 + lane)];
  __syncthreads();

  for (int t = 0; t < Tn; ++t) {
    if (t + 1 < Tn)
      __builtin_prefetch(&l_data[((size_t)b * Tn + (t + 1)) * LECn], 0, 1);

    // ---------------- question step ----------------
    const int qi  = q_data [(size_t)b * Tn + t];
    const int qai = qa_data[(size_t)b * Tn + t];
    if (tid < 128) {
      sQe[tid] = q_emb [(size_t)qi  * Kn + tid];
      sQa[tid] = qa_emb[(size_t)qai * Vn + tid];
      sQr[tid] = 0.0f; sLr[tid] = 0.0f;
    }
    __syncthreads();

    // cw = softmax(qe @ key^T): one score per thread, tree softmax in LDS
    float sc = -1e30f;
    if (tid < 128) {
      float s = 0.0f;
      for (int k = 0; k < 128; ++k) s = fmaf(sQe[k], (float)sWK[fragOff(k, tid)], s);
      sc = s;
    }
    sRed[tid] = sc;
    __syncthreads();
    for (int off = 128; off > 0; off >>= 1) {
      if (tid < off) sRed[tid] = fmaxf(sRed[tid], sRed[tid + off]);
      __syncthreads();
    }
    const float mx = sRed[0];
    __syncthreads();
    const float ex = (tid < 128) ? __expf(sc - mx) : 0.0f;
    sRed[tid] = ex;
    __syncthreads();
    for (int off = 128; off > 0; off >>= 1) {
      if (tid < off) sRed[tid] += sRed[tid + off];
      __syncthreads();
    }
    if (tid < 128) sCw[tid] = ex / sRed[0];
    __syncthreads();

    // q_read = cw . vm  (pre-update), cross-wave reduce via ds_add_f32
    {
      float p[4] = {0, 0, 0, 0};
      #pragma unroll
      for (int r = 0; r < 16; ++r) {
        const float wr = sCw[wv * 16 + r];
        #pragma unroll
        for (int q = 0; q < 4; ++q) p[q] = fmaf(wr, vm[r][q], p[q]);
      }
      #pragma unroll
      for (int q = 0; q < 4; ++q) atomicAdd(&sQr[q * 32 + lane], p[q]);
    }

    // e = sigmoid(qae@We^T+b), a = tanh(qae@Wa^T+b)  (single row -> VALU matvec)
    if (tid < 128) {
      float se = erase_b[tid], sa = add_b[tid];
      for (int k = 0; k < 128; ++k) {
        const float x = sQa[k];
        se = fmaf(x, (float)sWE[fragOff(k, tid)], se);
        sa = fmaf(x, (float)sWA[fragOff(k, tid)], sa);
      }
      sEq[tid] = 1.0f / (1.0f + __expf(-se));
      sAq[tid] = tanhf(sa);
    }
    __syncthreads();

    // vm = vm*(1 - cw⊗e) + cw⊗a   (register-resident rank-1 update)
    {
      float ev[4], av[4];
      #pragma unroll
      for (int q = 0; q < 4; ++q) { ev[q] = sEq[q * 32 + lane]; av[q] = sAq[q * 32 + lane]; }
      #pragma unroll
      for (int r = 0; r < 16; ++r) {
        const float wr = sCw[wv * 16 + r];
        #pragma unroll
        for (int q = 0; q < 4; ++q)
          vm[r][q] = fmaf(vm[r][q], 1.0f - wr * ev[q], wr * av[q]);
      }
    }

    // ---------------- LEC block: stage 128 l-embeddings ----------------
    if (tid < 128) sIdx[tid] = l_data[((size_t)b * Tn + t) * LECn + tid];
    __syncthreads();
    for (int i = tid; i < 128 * 128; i += 256) {
      const int j = i >> 7, k = i & 127;
      sX[j * HS + k] = (_Float16)l_emb[(size_t)sIdx[j] * Vn + k];
    }
    __syncthreads();
    if (tid < 128) {                       // ls = sum_j l_j
      float s = 0.0f;
      for (int j = 0; j < 128; ++j) s += (float)sX[j * HS + tid];
      sLs[tid] = s;
    }

    // Batched GEMMs on the matrix cores: scores, then fused E/A
    gemm_score(sX, sWK, sLcw, wv, lane);
    __syncthreads();
    if (tid < 128) {                       // row softmax of scores
      float m2 = -1e30f;
      for (int c = 0; c < 128; ++c) m2 = fmaxf(m2, sLcw[tid * FS + c]);
      float s2 = 0.0f;
      for (int c = 0; c < 128; ++c) {
        const float e = __expf(sLcw[tid * FS + c] - m2);
        sLcw[tid * FS + c] = e; s2 += e;
      }
      const float inv = 1.0f / s2;
      for (int c = 0; c < 128; ++c) sLcw[tid * FS + c] *= inv;
    }
    gemm_ea(sX, sWE, sWA, erase_b, add_b, sEA, wv, lane);
    __syncthreads();

    // Irreducible sequential rank-1 chain: 128 steps, vm in VGPRs, wide LDS reads
    const unsigned* __restrict__ pEA = (const unsigned*)sEA;
    float lrp[4] = {0, 0, 0, 0};
    for (int j = 0; j < LECn; ++j) {
      float ev[4], av[4];
      #pragma unroll
      for (int q = 0; q < 4; ++q) {
        EAu ea; ea.u = pEA[j * EAS + q * 32 + lane];   // one ds_load_b32 -> {e,a}
        ev[q] = (float)ea.h[0];
        av[q] = (float)ea.h[1];
      }
      alignas(16) float wl[16];
      const float4* w4 = (const float4*)(sLcw + j * FS + wv * 16);
      #pragma unroll
      for (int u = 0; u < 4; ++u) *(float4*)(wl + 4 * u) = w4[u];  // 4x ds_load_b128
      #pragma unroll
      for (int r = 0; r < 16; ++r) {
        const float wr = wl[r];
        #pragma unroll
        for (int q = 0; q < 4; ++q) {
          lrp[q]   = fmaf(wr, vm[r][q], lrp[q]);              // read before write
          vm[r][q] = fmaf(vm[r][q], 1.0f - wr * ev[q], wr * av[q]);
        }
      }
    }
    #pragma unroll
    for (int q = 0; q < 4; ++q) atomicAdd(&sLr[q * 32 + lane], lrp[q]);
    __syncthreads();

    // ---------------- summary head ----------------
    if (tid < 128) {
      sM[tid]       = sQr[tid];
      sM[128 + tid] = sQe[tid];
      sM[256 + tid] = sLr[tid];
      sM[384 + tid] = sLs[tid];
    }
    __syncthreads();
    {
      float acc = summary_b[tid];
      const float* Wrow = &summary_W[(size_t)tid * (2 * Kn + 2 * Vn)];
      for (int i = 0; i < 2 * Kn + 2 * Vn; ++i) acc = fmaf(sM[i], Wrow[i], acc);
      sH[tid] = tanhf(acc);
    }
    __syncthreads();
    if (tid == 0) {
      float acc = out_b[0];
      for (int o = 0; o < Sn; ++o) acc = fmaf(sH[o], out_W[o], acc);
      out[(size_t)b * Tn + t] = 1.0f / (1.0f + __expf(-acc));
    }
    __syncthreads();   // buffers reused next t
  }
}

extern "C" void kernel_launch(void* const* d_in, const int* in_sizes, int n_in,
                              void* d_out, int out_size, void* d_ws, size_t ws_size,
                              hipStream_t stream) {
  (void)in_sizes; (void)n_in; (void)d_ws; (void)ws_size; (void)out_size;
  const int*   q_data     = (const int*)  d_in[0];
  const int*   qa_data    = (const int*)  d_in[1];
  const int*   l_data     = (const int*)  d_in[2];
  const float* key_matrix = (const float*)d_in[3];
  const float* value_init = (const float*)d_in[4];
  const float* q_emb      = (const float*)d_in[5];
  const float* qa_emb     = (const float*)d_in[6];
  const float* l_emb      = (const float*)d_in[7];
  const float* erase_W    = (const float*)d_in[8];
  const float* erase_b    = (const float*)d_in[9];
  const float* add_W      = (const float*)d_in[10];
  const float* add_b      = (const float*)d_in[11];
  const float* summary_W  = (const float*)d_in[12];
  const float* summary_b  = (const float*)d_in[13];
  const float* out_W      = (const float*)d_in[14];
  const float* out_b      = (const float*)d_in[15];
  float* outp = (float*)d_out;

  hipFuncSetAttribute((const void*)dmkt_kernel,
                      hipFuncAttributeMaxDynamicSharedMemorySize, (int)SMEM_TOTAL);
  dmkt_kernel<<<Bn, 256, SMEM_TOTAL, stream>>>(
      q_data, qa_data, l_data, key_matrix, value_init, q_emb, qa_emb, l_emb,
      erase_W, erase_b, add_W, add_b, summary_W, summary_b, out_W, out_b, outp);
}
